// GlobalAttentionPool_7576322310464
// MI455X (gfx1250) — matrix operand order repaired and checked by reading, not compiled, not run on hardware
//
#include <hip/hip_runtime.h>
#include <hip/hip_bf16.h>
#include <math.h>

typedef __attribute__((ext_vector_type(16))) _Float16 v16h;
typedef __attribute__((ext_vector_type(8)))  float    v8f;
typedef __attribute__((ext_vector_type(4)))  float    f32x4;
typedef unsigned int u32;
typedef __attribute__((ext_vector_type(4))) u32 u32x4;
typedef __attribute__((ext_vector_type(8))) int i32x8;
typedef __attribute__((ext_vector_type(4))) int i32x4;

#define HDIM   256
#define HHALF  128
#define MTILE  128   // x rows per workgroup (8 waves x 16 rows)

#if __has_builtin(__builtin_amdgcn_tensor_load_to_lds) && \
    __has_builtin(__builtin_amdgcn_s_wait_tensorcnt)
#define USE_TDM 1
#else
#define USE_TDM 0
#endif

// ---------------------------------------------------------------------------
// Kernel 1: scores[n] = tanh(x[n,:] @ W1 + b1) @ W2 + b2   via WMMA f16
// x tile staged raw (f32) into LDS by the Tensor Data Mover; W1 staged
// transposed as f16 by the waves (transpose is not a TDM-friendly pattern).
// ---------------------------------------------------------------------------
__global__ __launch_bounds__(256) void score_kernel(
    const float* __restrict__ x,  const float* __restrict__ W1,
    const float* __restrict__ b1, const float* __restrict__ W2,
    const float* __restrict__ b2, float* __restrict__ scores, int nTiles)
{
    extern __shared__ float smemf[];
    float*    xt32 = smemf;                          // [MTILE][HDIM] f32 (128 KB)
    _Float16* wt   = (_Float16*)(smemf + MTILE * HDIM); // [HHALF][HDIM] f16 (64 KB)

    const int tid  = threadIdx.x;
    const int tile = blockIdx.x;
    const long long rowBaseG = (long long)tile * MTILE;
    const float* xg = x + rowBaseG * HDIM;

    // Prefetch next tile of x (lowers to global_prefetch_b8).
    if (tile + 1 < nTiles)
        __builtin_prefetch(x + (rowBaseG + MTILE) * HDIM + tid * 4, 0, 0);

    // ---- stage x tile (raw f32, fully contiguous 128 KB) into LDS ----
#if USE_TDM
    if (tid < 32) {  // wave 0 issues a single TDM copy for the whole tile
        const unsigned long long ga = (unsigned long long)(const void*)xg;
        // LDS aperture lives entirely in the high 32 bits of a flat address,
        // so the low 32 bits of a generic LDS pointer are the LDS byte offset.
        const u32 ldsAddr = (u32)(unsigned long long)(const void*)xt32;
        const u32 ELEMS = MTILE * HDIM;  // 32768 x 4-byte elements

        u32x4 g0;
        g0.x = 1u;                                           // count=1 (valid D#)
        g0.y = ldsAddr;                                      // lds_addr[31:0]
        g0.z = (u32)(ga & 0xffffffffu);                      // global_addr[31:0]
        g0.w = (u32)((ga >> 32) & 0x01ffffffu) | (2u << 30); // global_addr[56:32], type=2

        i32x8 g1;
        g1[0] = (int)(2u << 16);                      // wg_mask=0, data_size=2 (4 B)
        g1[1] = (int)((ELEMS & 0xffffu) << 16);       // tensor_dim0[15:0]
        g1[2] = (int)(((ELEMS >> 16) & 0xffffu) | (1u << 16)); // tensor_dim0[31:16], tensor_dim1=1
        g1[3] = (int)((ELEMS & 0xffffu) << 16);       // tile_dim0 = 32768
        g1[4] = 1;                                    // tile_dim1=1, tile_dim2=0
        g1[5] = (int)ELEMS;                           // tensor_dim0_stride[31:0]
        g1[6] = 0;
        g1[7] = 0;

        i32x4 g2 = {0, 0, 0, 0};
        i32x4 g3 = {0, 0, 0, 0};
#if __clang_major__ >= 23
        i32x8 g4 = {0, 0, 0, 0, 0, 0, 0, 0};
        __builtin_amdgcn_tensor_load_to_lds(g0, g1, g2, g3, g4, 0);
#else
        __builtin_amdgcn_tensor_load_to_lds(g0, g1, g2, g3, 0);
#endif
    }
#else
    #pragma unroll
    for (int it = 0; it < (MTILE * HDIM / 4) / 256; ++it) {
        const int i4 = tid + it * 256;
        ((f32x4*)xt32)[i4] = ((const f32x4*)xg)[i4];
    }
#endif

    // ---- stage W1 transposed: wt[j][k] = W1[k][j], coalesced LDS writes ----
    for (int it = 0; it < (HHALF * HDIM) / 256; ++it) {
        const int idx = tid + it * 256;      // linear index into wt
        const int j = idx / HDIM;            // output column 0..127
        const int k = idx % HDIM;            // reduction dim 0..255
        wt[idx] = (_Float16)W1[k * HHALF + j];
    }

#if USE_TDM
    if (tid < 32) __builtin_amdgcn_s_wait_tensorcnt(0);
#endif
    __syncthreads();

    // ---- per-wave 16x128 tile of h = xt @ W1 ----
    const int wave = tid >> 5;
    const int lane = tid & 31;
    const int half = lane >> 4;   // 0: lanes 0-15, 1: lanes 16-31
    const int lm   = lane & 15;

    // A-fragment row for this lane (16-bit A 16x32 layout: M = lane%16)
    const float* xr32 = xt32 + (wave * 16 + lm) * HDIM;

    v8f acc[8] = {};  // 8 col-blocks of 16 cols, 16x16 f32 accum each

    #pragma unroll
    for (int ks = 0; ks < 8; ++ks) {          // K steps of 32 (K = 256)
        // A fragment: lane holds K = ks*32 + half*16 + {0..15}
        const int k0 = ks * 32 + half * 16;
        const f32x4 p0 = *(const f32x4*)(xr32 + k0);
        const f32x4 p1 = *(const f32x4*)(xr32 + k0 + 4);
        const f32x4 p2 = *(const f32x4*)(xr32 + k0 + 8);
        const f32x4 p3 = *(const f32x4*)(xr32 + k0 + 12);
        v16h a;
        #pragma unroll
        for (int i = 0; i < 4; ++i) {
            a[i]      = (_Float16)p0[i];
            a[4 + i]  = (_Float16)p1[i];
            a[8 + i]  = (_Float16)p2[i];
            a[12 + i] = (_Float16)p3[i];
        }

        #pragma unroll
        for (int c = 0; c < 8; ++c) {         // 8 blocks of 16 output cols
            // B fragment (32x16): lane = col c*16+lm, K = ks*32 + half*16 + {0..15}
            const _Float16* wr = wt + (c * 16 + lm) * HDIM + ks * 32 + half * 16;
            const v16h bfrag = *(const v16h*)wr;
            acc[c] = __builtin_amdgcn_wmma_f32_16x16x32_f16(
                false, a, false, bfrag, (short)0, acc[c], false, false);
        }
    }

    // ---- epilogue: tanh, dot with W2, reduce across 16-lane halves ----
    // C layout: lane holds col = c*16 + lm, rows j + 8*half (j = 0..7)
    float partial[8] = {0.f,0.f,0.f,0.f,0.f,0.f,0.f,0.f};
    #pragma unroll
    for (int c = 0; c < 8; ++c) {
        const int col = c * 16 + lm;
        const float bb = b1[col];
        const float w2 = W2[col];
        #pragma unroll
        for (int j = 0; j < 8; ++j)
            partial[j] += tanhf(acc[c][j] + bb) * w2;
    }
    #pragma unroll
    for (int j = 0; j < 8; ++j) {
        float v = partial[j];
        #pragma unroll
        for (int off = 8; off; off >>= 1)     // stays within each 16-lane half
            v += __shfl_xor(v, off, 32);
        partial[j] = v;
    }
    if (lm == 0) {
        const long long r0 = rowBaseG + wave * 16 + half * 8;
        const float b2v = b2[0];
        #pragma unroll
        for (int j = 0; j < 8; ++j)
            scores[r0 + j] = partial[j] + b2v;
    }
}

// ---------------------------------------------------------------------------
// Kernel 2: segment softmax + weighted pooling. batch_idx sorted -> one block
// per group, binary search for bounds, no atomics.
// ---------------------------------------------------------------------------
__global__ __launch_bounds__(256) void pool_kernel(
    const float* __restrict__ x, const int* __restrict__ bidx,
    const float* __restrict__ scores, float* __restrict__ out, int N)
{
    const int g   = blockIdx.x;
    const int tid = threadIdx.x;
    __shared__ float red[256];
    __shared__ float wbuf[256];
    __shared__ int   sb[2];
    __shared__ float sv[2];

    if (tid < 2) {  // lower_bound for g and g+1
        const int target = g + tid;
        int lo = 0, hi = N;
        while (lo < hi) {
            const int mid = (lo + hi) >> 1;
            if (bidx[mid] < target) lo = mid + 1; else hi = mid;
        }
        sb[tid] = lo;
    }
    __syncthreads();
    const int start = sb[0], end = sb[1];

    // segment max
    float m = -INFINITY;
    for (int i = start + tid; i < end; i += 256) m = fmaxf(m, scores[i]);
    red[tid] = m; __syncthreads();
    for (int s = 128; s; s >>= 1) {
        if (tid < s) red[tid] = fmaxf(red[tid], red[tid + s]);
        __syncthreads();
    }
    if (tid == 0) sv[0] = red[0];
    __syncthreads();
    const float smax = sv[0];

    // segment sum of exp
    float se = 0.f;
    for (int i = start + tid; i < end; i += 256) se += expf(scores[i] - smax);
    red[tid] = se; __syncthreads();
    for (int s = 128; s; s >>= 1) {
        if (tid < s) red[tid] += red[tid + s];
        __syncthreads();
    }
    if (tid == 0) sv[1] = 1.0f / (red[0] + 1e-16f);
    __syncthreads();
    const float invSum = sv[1];

    // weighted accumulation: thread owns one of the 256 columns
    float accv = 0.f;
    for (int base = start; base < end; base += 256) {
        const int node = base + tid;
        wbuf[tid] = (node < end) ? expf(scores[node] - smax) * invSum : 0.f;
        __syncthreads();
        const int cnt = min(256, end - base);
        for (int i = 0; i < cnt; ++i)
            accv += x[(long long)(base + i) * HDIM + tid] * wbuf[i];
        __syncthreads();
    }
    out[(long long)g * HDIM + tid] = accv;
}

// ---------------------------------------------------------------------------
extern "C" void kernel_launch(void* const* d_in, const int* in_sizes, int n_in,
                              void* d_out, int out_size, void* d_ws, size_t ws_size,
                              hipStream_t stream) {
    const float* x    = (const float*)d_in[0];
    const int*   bidx = (const int*)  d_in[1];
    const float* W1   = (const float*)d_in[2];
    const float* b1   = (const float*)d_in[3];
    const float* W2   = (const float*)d_in[4];
    const float* b2   = (const float*)d_in[5];
    float* out = (float*)d_out;

    const int N = in_sizes[0] / HDIM;   // 524288
    const int G = out_size / HDIM;      // 4096
    float* scores = (float*)d_ws;       // N floats of scratch

    const int nTiles = N / MTILE;       // 4096
    // 128 KB f32 x-tile + 64 KB f16 W1^T = 192 KB of the WGP's 320 KB LDS
    const size_t lds = (size_t)(MTILE * HDIM) * sizeof(float)
                     + (size_t)(HHALF * HDIM) * sizeof(_Float16);

    score_kernel<<<nTiles, 256, lds, stream>>>(x, W1, b1, W2, b2, scores, nTiles);
    pool_kernel<<<G, 256, 0, stream>>>(x, bidx, scores, out, N);
}
